// GCNEncoder_39960375722252
// MI455X (gfx1250) — matrix operand order repaired and checked by reading, not compiled, not run on hardware
//
#include <hip/hip_runtime.h>

typedef __attribute__((ext_vector_type(16))) _Float16 v16h;
typedef __attribute__((ext_vector_type(8)))  float    v8f;

#define N_NODES 50000
#define N_EDGES 800000
#define F_HID   128
#define F_OUT   64

// ---------------------------------------------------------------------------
// Fragment-layout index helpers (CDNA5 wave32 WMMA 16x16x32 f16 layouts)
// A (16x32 MxK): lanes 0-15 -> M=lane, K in {0..7,16..23}; lanes 16-31 -> K in {8..15,24..31}
// B (32x16 KxN): lanes 0-15 -> N=lane, K in {0..7,16..23}; lanes 16-31 -> K in {8..15,24..31}
// ---------------------------------------------------------------------------
__device__ __forceinline__ size_t a_pack_idx(int row, int col) {
    int rt = row >> 4, lr = row & 15;
    int kk = col >> 5, rem = col & 31;
    int laneHi = (rem >> 3) & 1;                     // which half-wave holds this K
    int h      = (rem & 7) | (((rem >> 4) & 1) << 3); // half index within 16
    int lane   = (laneHi << 4) | lr;
    return ((((size_t)rt * 4 + kk) * 32 + lane) << 4) + h;
}

__device__ __forceinline__ size_t b_pack_idx(int k, int col) {
    int ct = col >> 4, lc = col & 15;
    int kk = k >> 5, kr = k & 31;
    int laneHi = (kr >> 3) & 1;
    int h      = (kr & 7) | (((kr >> 4) & 1) << 3);
    int lane   = (laneHi << 4) | lc;
    return ((((size_t)ct * 4 + kk) * 32 + lane) << 4) + h;
}

// ---------------------------------------------------------------------------
// Degree / normalization
// ---------------------------------------------------------------------------
__global__ void k_deg(const int* __restrict__ dst, unsigned* deg, int e_count) {
    int e = blockIdx.x * blockDim.x + threadIdx.x;
    if (e < e_count) atomicAdd(&deg[dst[e]], 1u);
}

__global__ void k_dinv(unsigned* deg_in, float* dinv_out, int n) {
    int i = blockIdx.x * blockDim.x + threadIdx.x;
    if (i < n) {
        float d = (float)(deg_in[i] + 1u);  // +1 self loop
        dinv_out[i] = rsqrtf(d);            // in-place over same buffer is safe
    }
}

// ---------------------------------------------------------------------------
// Pack fp32 activations [n,128] -> f16 A-fragment layout
// ---------------------------------------------------------------------------
__global__ void k_pack_a(const float* __restrict__ x, _Float16* __restrict__ apack, int total) {
    int t = blockIdx.x * blockDim.x + threadIdx.x;
    if (t >= total) return;
    int row = t >> 7, col = t & 127;
    apack[a_pack_idx(row, col)] = (_Float16)x[t];
}

// Pack fp32 weights [128,Fout] -> f16 B-fragment layout
__global__ void k_pack_b(const float* __restrict__ w, _Float16* __restrict__ bpack,
                         int fout, int total) {
    int t = blockIdx.x * blockDim.x + threadIdx.x;
    if (t >= total) return;
    int k = t / fout, col = t - k * fout;
    bpack[b_pack_idx(k, col)] = (_Float16)w[t];
}

// ---------------------------------------------------------------------------
// WMMA GEMM: [16 rows/block] x [Fout cols], K=128. blockDim=(32, Fout/16).
// Optional fused bias+ReLU epilogue (used for the final layer).
// ---------------------------------------------------------------------------
__global__ void k_gemm(const _Float16* __restrict__ apack,
                       const _Float16* __restrict__ bpack,
                       float* __restrict__ out, int fout,
                       const float* __restrict__ bias, int relu) {
    const int lane = threadIdx.x;      // 0..31
    const int ct   = threadIdx.y;      // column tile
    const int rt   = blockIdx.x;       // row tile

    const v16h* A = reinterpret_cast<const v16h*>(apack) + ((size_t)rt * 4) * 32 + lane;
    const v16h* B = reinterpret_cast<const v16h*>(bpack) + ((size_t)ct * 4) * 32 + lane;

    v8f acc = {};
#pragma unroll
    for (int kk = 0; kk < 4; ++kk) {
        v16h a = A[kk * 32];
        v16h b = B[kk * 32];
        acc = __builtin_amdgcn_wmma_f32_16x16x32_f16(
            false, a, false, b, (short)0, acc, false, false);
    }

    const int col  = ct * 16 + (lane & 15);
    const int row0 = rt * 16 + ((lane >> 4) << 3);  // lanes 16-31 hold M+8
    float bv = bias ? bias[col] : 0.0f;
#pragma unroll
    for (int r = 0; r < 8; ++r) {
        float v = acc[r] + bv;
        if (relu) v = fmaxf(v, 0.0f);
        out[(size_t)(row0 + r) * fout + col] = v;
    }
}

// ---------------------------------------------------------------------------
// Edge scatter: one wave32 per edge; each lane handles 4 features (32*4=128).
// src/dst/norm scalarized via readfirstlane -> s_loads; payload float4 loads;
// native f32 global atomic adds into agg.
// ---------------------------------------------------------------------------
__global__ void k_scatter(const int* __restrict__ src, const int* __restrict__ dst,
                          const float* __restrict__ dinv, const float* __restrict__ h,
                          float* __restrict__ agg, int e_count) {
    int e = blockIdx.x * (blockDim.x >> 5) + (threadIdx.x >> 5);
    e = __builtin_amdgcn_readfirstlane(e);
    if (e >= e_count) return;
    int s = __builtin_amdgcn_readfirstlane(src[e]);
    int d = __builtin_amdgcn_readfirstlane(dst[e]);
    float w = dinv[s] * dinv[d];

    int f4 = (threadIdx.x & 31) << 2;
    const float4 m = *reinterpret_cast<const float4*>(h + (size_t)s * F_HID + f4);
    float* base = agg + (size_t)d * F_HID + f4;
    unsafeAtomicAdd(base + 0, m.x * w);
    unsafeAtomicAdd(base + 1, m.y * w);
    unsafeAtomicAdd(base + 2, m.z * w);
    unsafeAtomicAdd(base + 3, m.w * w);
}

// ---------------------------------------------------------------------------
// Finalize layer: add self-loop contribution h*dinv^2, bias, ReLU; emit the
// result packed as f16 A-fragments for the next GEMM.
// ---------------------------------------------------------------------------
__global__ void k_finalize(const float* __restrict__ agg, const float* __restrict__ h,
                           const float* __restrict__ dinv, const float* __restrict__ bias,
                           _Float16* __restrict__ apack_next, int total) {
    int t = blockIdx.x * blockDim.x + threadIdx.x;
    if (t >= total) return;
    int row = t >> 7, col = t & 127;
    float di = dinv[row];
    float v = agg[t] + h[t] * di * di + bias[col];
    v = fmaxf(v, 0.0f);
    apack_next[a_pack_idx(row, col)] = (_Float16)v;
}

// ---------------------------------------------------------------------------
// Host launch
// ---------------------------------------------------------------------------
extern "C" void kernel_launch(void* const* d_in, const int* in_sizes, int n_in,
                              void* d_out, int out_size, void* d_ws, size_t ws_size,
                              hipStream_t stream) {
    const float* x   = (const float*)d_in[0];
    const int*   ei  = (const int*)d_in[1];   // [2,E] row-major: src then dst
    const float* W1  = (const float*)d_in[2];
    const float* b1  = (const float*)d_in[3];
    const float* W2  = (const float*)d_in[4];
    const float* b2  = (const float*)d_in[5];
    const float* W3  = (const float*)d_in[6];
    const float* b3  = (const float*)d_in[7];
    float* out = (float*)d_out;

    const int n = N_NODES, e_count = N_EDGES;
    const int* src = ei;
    const int* dst = ei + e_count;

    // Workspace carve-up (256B aligned)
    char* ws = (char*)d_ws;
    size_t off = 0;
    auto alloc = [&](size_t bytes) {
        char* p = ws + off;
        off += (bytes + 255) & ~(size_t)255;
        return p;
    };
    float*    dinv  = (float*)alloc((size_t)n * 4);                // also deg (u32) scratch
    _Float16* apack = (_Float16*)alloc((size_t)n * F_HID * 2);
    _Float16* bpack = (_Float16*)alloc((size_t)F_HID * F_HID * 2);
    float*    h     = (float*)alloc((size_t)n * F_HID * 4);
    float*    agg   = (float*)alloc((size_t)n * F_HID * 4);
    (void)ws_size;

    const int TPB = 256;
    const int rowTiles = n / 16;                       // 3125 exact
    const size_t actTotal = (size_t)n * F_HID;         // 6.4M elems
    const int actBlocks = (int)((actTotal + TPB - 1) / TPB);
    const int scatBlocks = (e_count * 32 + TPB - 1) / TPB;  // 1 wave per edge

    // --- normalization: deg -> dinv ---
    hipMemsetAsync(dinv, 0, (size_t)n * 4, stream);
    k_deg<<<(e_count + TPB - 1) / TPB, TPB, 0, stream>>>(dst, (unsigned*)dinv, e_count);
    k_dinv<<<(n + TPB - 1) / TPB, TPB, 0, stream>>>((unsigned*)dinv, dinv, n);

    // --- layer 1 ---
    k_pack_a<<<actBlocks, TPB, 0, stream>>>(x, apack, (int)actTotal);
    k_pack_b<<<(F_HID * F_HID + TPB - 1) / TPB, TPB, 0, stream>>>(W1, bpack, F_HID, F_HID * F_HID);
    k_gemm<<<rowTiles, dim3(32, F_HID / 16), 0, stream>>>(apack, bpack, h, F_HID, nullptr, 0);
    hipMemsetAsync(agg, 0, actTotal * 4, stream);
    k_scatter<<<scatBlocks, TPB, 0, stream>>>(src, dst, dinv, h, agg, e_count);
    k_finalize<<<actBlocks, TPB, 0, stream>>>(agg, h, dinv, b1, apack, (int)actTotal);

    // --- layer 2 ---
    k_pack_b<<<(F_HID * F_HID + TPB - 1) / TPB, TPB, 0, stream>>>(W2, bpack, F_HID, F_HID * F_HID);
    k_gemm<<<rowTiles, dim3(32, F_HID / 16), 0, stream>>>(apack, bpack, h, F_HID, nullptr, 0);
    hipMemsetAsync(agg, 0, actTotal * 4, stream);
    k_scatter<<<scatBlocks, TPB, 0, stream>>>(src, dst, dinv, h, agg, e_count);
    k_finalize<<<actBlocks, TPB, 0, stream>>>(agg, h, dinv, b2, apack, (int)actTotal);

    // --- layer 3: GEMM with fused bias + ReLU straight into d_out ---
    k_pack_b<<<(F_HID * F_OUT + TPB - 1) / TPB, TPB, 0, stream>>>(W3, bpack, F_OUT, F_HID * F_OUT);
    k_gemm<<<rowTiles, dim3(32, F_OUT / 16), 0, stream>>>(apack, bpack, out, F_OUT, b3, 1);
}
